// AttentionLocal_31078383354536
// MI455X (gfx1250) — matrix operand order repaired and checked by reading, not compiled, not run on hardware
//
#include <hip/hip_runtime.h>

// ---------------------------------------------------------------------------
// Problem constants (from reference): B=64, C=256, H=W=LF=32, C1=256, LF^2=1024
// ---------------------------------------------------------------------------
#define BATCH   64
#define CIN     256
#define HW      1024          // 32*32
#define C1      256
#define C2      1024          // LF*LF
#define NPIX    (BATCH * HW)  // 65536 GEMM rows

typedef __attribute__((ext_vector_type(16))) __bf16        v16bf;
typedef __attribute__((ext_vector_type(8)))  float         v8f;
typedef __attribute__((ext_vector_type(4)))  unsigned int  v4u;

// LDS tile geometry for the WMMA GEMMs
#define BM 128
#define BN 64
#define BK 32
#define LDS_STRIDE 40   // bf16 elems per LDS row (80B, 16B-aligned, conflict-padded)

union FragU { v16bf bf; v4u u[2]; };

// A-matrix fragment (16x32 bf16): lanes 0-15 hold K[0..7]+K[16..23] of row m,
// lanes 16-31 hold K[8..15]+K[24..31] of row m (ISA 7.12.2 16-bit A layout).
__device__ inline v16bf load_frag_A(const __bf16* tile, int m, int lane) {
    const int half = lane >> 4;
    const v4u* p = reinterpret_cast<const v4u*>(tile + m * LDS_STRIDE);
    FragU r;
    r.u[0] = p[half];       // K = half*8 .. half*8+7
    r.u[1] = p[2 + half];   // K = 16+half*8 .. 16+half*8+7
    return r.bf;
}

// B-matrix fragment (32x16 bf16), tile stored [n][k]:
// lanes 0-15 hold K[0..7]+K[16..23] of col n, lanes 16-31 hold K[8..15]+K[24..31].
__device__ inline v16bf load_frag_B(const __bf16* tile, int n, int lane) {
    const int half = lane >> 4;
    const v4u* p = reinterpret_cast<const v4u*>(tile + n * LDS_STRIDE);
    FragU r;
    r.u[0] = p[half];
    r.u[1] = p[2 + half];
    return r.bf;
}

// ---------------------------------------------------------------------------
// Layout prep kernels (fp32 -> bf16, optional transposes)
// ---------------------------------------------------------------------------

// Plain fp32 -> bf16 convert (same layout). Used for xN [B][C][HW] and w2 [oc2][c].
__global__ void cvt_bf16_kernel(const float* __restrict__ src, __bf16* __restrict__ dst) {
    long long i = (long long)blockIdx.x * blockDim.x + threadIdx.x;
    dst[i] = (__bf16)src[i];
}

// x [B][C][HW] fp32 -> xT [B][HW][C] bf16 (conv1's gathered A operand)
__global__ void prep_xT_kernel(const float* __restrict__ x, __bf16* __restrict__ xT) {
    long long i = (long long)blockIdx.x * blockDim.x + threadIdx.x;  // [b][hw][c]
    int c  = (int)(i & 255);
    int hw = (int)((i >> 8) & 1023);
    int b  = (int)(i >> 18);
    xT[i] = (__bf16)x[((long long)b * CIN + c) * HW + hw];
}

// conv1_w [oc][c][7][7] fp32 -> w1t [tap][oc][c] bf16  (B^T layout per tap)
__global__ void prep_w1t_kernel(const float* __restrict__ w, __bf16* __restrict__ wt) {
    long long i = (long long)blockIdx.x * blockDim.x + threadIdx.x;  // [tap][oc][c]
    int c   = (int)(i & 255);
    int oc  = (int)((i >> 8) & 255);
    int tap = (int)(i >> 16);
    wt[i] = (__bf16)w[((long long)oc * CIN + c) * 49 + tap];
}

// ---------------------------------------------------------------------------
// Generic bf16 WMMA GEMM: A[M][K] row-major, Bt[N][K] row-major (= B^T),
// C[M][N] (+optional bias), batched over blockIdx.z via element strides.
// M%128==0, N%64==0, K%32==0. Ping-pong LDS double buffering, 1 barrier/chunk.
// ---------------------------------------------------------------------------
template <typename OutT>
__global__ __launch_bounds__(256)
void gemm_bf16_kernel(const __bf16* __restrict__ A, const __bf16* __restrict__ Bt,
                      OutT* __restrict__ C, const float* __restrict__ bias,
                      int M, int N, int K,
                      long long aBatch, long long bBatch, long long cBatch)
{
    __shared__ __bf16 As[2][BM * LDS_STRIDE];
    __shared__ __bf16 Bs[2][BN * LDS_STRIDE];

    const int tid  = threadIdx.x;
    const int lane = tid & 31;
    const int wave = tid >> 5;
    const int wm   = wave >> 1;      // 0..3 (M dir)
    const int wn   = wave & 1;       // 0..1 (N dir)

    const long long z = blockIdx.z;
    A  += z * aBatch;
    Bt += z * bBatch;
    C  += z * cBatch;

    const int m0 = blockIdx.y * BM;
    const int n0 = blockIdx.x * BN;

    // per-thread staging coords: A row 0..127 (16 elems), B row 0..63 (8 elems)
    const int arow = tid >> 1, akh = (tid & 1) * 16;
    const int brow = tid >> 2, bq  = (tid & 3) * 8;
    const __bf16* aBase = A  + (long long)(m0 + arow) * K + akh;
    const __bf16* bBase = Bt + (long long)(n0 + brow) * K + bq;

    const int nk = K / BK;
    v4u ra0, ra1, rb;
    {   // preload chunk 0 and stage into buffer 0
        const v4u* s = reinterpret_cast<const v4u*>(aBase);
        ra0 = s[0]; ra1 = s[1];
        rb = *reinterpret_cast<const v4u*>(bBase);
        v4u* da = reinterpret_cast<v4u*>(&As[0][arow * LDS_STRIDE + akh]);
        da[0] = ra0; da[1] = ra1;
        *reinterpret_cast<v4u*>(&Bs[0][brow * LDS_STRIDE + bq]) = rb;
    }
    __syncthreads();

    v8f acc[2][2] = {};
    int buf = 0;

    for (int kc = 0; kc < nk; ++kc) {
        const bool more = (kc + 1) < nk;
        if (more) {   // issue global fetches for chunk kc+1 (overlap with WMMAs)
            const v4u* s = reinterpret_cast<const v4u*>(aBase + (kc + 1) * BK);
            ra0 = s[0]; ra1 = s[1];
            rb = *reinterpret_cast<const v4u*>(bBase + (kc + 1) * BK);
            if (kc + 2 < nk) {
                __builtin_prefetch(aBase + (kc + 2) * BK, 0, 1);  // global_prefetch_b8
                __builtin_prefetch(bBase + (kc + 2) * BK, 0, 1);
            }
        }

        // compute chunk kc from current buffer
        v16bf af[2], bfr[2];
        #pragma unroll
        for (int i = 0; i < 2; ++i)
            af[i] = load_frag_A(As[buf], wm * 32 + i * 16 + (lane & 15), lane);
        #pragma unroll
        for (int j = 0; j < 2; ++j)
            bfr[j] = load_frag_B(Bs[buf], wn * 32 + j * 16 + (lane & 15), lane);
        #pragma unroll
        for (int i = 0; i < 2; ++i)
            #pragma unroll
            for (int j = 0; j < 2; ++j)
                acc[i][j] = __builtin_amdgcn_wmma_f32_16x16x32_bf16(
                    false, af[i], false, bfr[j], (short)0, acc[i][j], false, false);

        if (more) {   // stage chunk kc+1 into the alternate buffer
            v4u* da = reinterpret_cast<v4u*>(&As[buf ^ 1][arow * LDS_STRIDE + akh]);
            da[0] = ra0; da[1] = ra1;
            *reinterpret_cast<v4u*>(&Bs[buf ^ 1][brow * LDS_STRIDE + bq]) = rb;
            buf ^= 1;
        }
        __syncthreads();
    }

    // store (ISA C/D layout: lane = N, VGPR v = M row (+8 for high lane half))
    const int half = lane >> 4, ln = lane & 15;
    #pragma unroll
    for (int i = 0; i < 2; ++i) {
        const int mb = m0 + wm * 32 + i * 16 + half * 8;
        #pragma unroll
        for (int j = 0; j < 2; ++j) {
            const int nb = n0 + wn * 32 + j * 16 + ln;
            const float bv = bias ? bias[nb] : 0.0f;
            #pragma unroll
            for (int v = 0; v < 8; ++v)
                C[(long long)(mb + v) * N + nb] = (OutT)(acc[i][j][v] + bv);
        }
    }
}

// ---------------------------------------------------------------------------
// conv1: implicit GEMM, M=65536 pixels x N=256 oc, K = 49 taps x 256 channels.
// A gathered from xT (dilation-2 7x7 window, zero-fill OOB); Bt = w1t[tap][oc][c].
// Same ping-pong double-buffered scheme, chunk id kc = tap*8 + c-block.
// ---------------------------------------------------------------------------
__global__ __launch_bounds__(256)
void conv1_wmma_kernel(const __bf16* __restrict__ xT,   // [B][HW][C]
                       const __bf16* __restrict__ w1t,  // [49][C1][C]
                       const float*  __restrict__ bias, // [C1]
                       float*        __restrict__ h1)   // [NPIX][C1]
{
    __shared__ __bf16 As[2][BM * LDS_STRIDE];
    __shared__ __bf16 Bs[2][BN * LDS_STRIDE];

    const int tid  = threadIdx.x;
    const int lane = tid & 31;
    const int wave = tid >> 5;
    const int wm   = wave >> 1;
    const int wn   = wave & 1;

    const int m0 = blockIdx.y * BM;  // pixel base (all 128 share one batch b)
    const int n0 = blockIdx.x * BN;  // oc base
    const int b  = m0 >> 10;
    const __bf16* xb = xT + (long long)b * HW * CIN;

    const int arow = tid >> 1, akh = (tid & 1) * 16;
    const int brow = tid >> 2, bq  = (tid & 3) * 8;
    const int hw = (m0 + arow) & 1023;
    const int y0 = hw >> 5, x0 = hw & 31;

    const int nk = 49 * (CIN / BK);  // 392 chunks
    v4u ra0, ra1, rb;

    auto fetch = [&](int kc) {
        const int tap = kc >> 3;
        const int c0  = (kc & 7) * BK;
        const int dy = (tap / 7) * 2 - 6;   // dilation 2, pad 6
        const int dx = (tap % 7) * 2 - 6;
        const int yy = y0 + dy, xx = x0 + dx;
        const bool valid = (yy >= 0) & (yy < 32) & (xx >= 0) & (xx < 32);
        v4u z = {0, 0, 0, 0};
        ra0 = z; ra1 = z;
        if (valid) {
            const v4u* s = reinterpret_cast<const v4u*>(
                xb + (long long)(yy * 32 + xx) * CIN + c0 + akh);
            ra0 = s[0]; ra1 = s[1];
        }
        rb = *reinterpret_cast<const v4u*>(
            w1t + (long long)tap * CIN * C1 + (long long)(n0 + brow) * CIN + c0 + bq);
    };

    fetch(0);
    {
        v4u* da = reinterpret_cast<v4u*>(&As[0][arow * LDS_STRIDE + akh]);
        da[0] = ra0; da[1] = ra1;
        *reinterpret_cast<v4u*>(&Bs[0][brow * LDS_STRIDE + bq]) = rb;
    }
    __syncthreads();

    v8f acc[2][2] = {};
    int buf = 0;

    for (int kc = 0; kc < nk; ++kc) {
        const bool more = (kc + 1) < nk;
        if (more) fetch(kc + 1);

        v16bf af[2], bfr[2];
        #pragma unroll
        for (int i = 0; i < 2; ++i)
            af[i] = load_frag_A(As[buf], wm * 32 + i * 16 + (lane & 15), lane);
        #pragma unroll
        for (int j = 0; j < 2; ++j)
            bfr[j] = load_frag_B(Bs[buf], wn * 32 + j * 16 + (lane & 15), lane);
        #pragma unroll
        for (int i = 0; i < 2; ++i)
            #pragma unroll
            for (int j = 0; j < 2; ++j)
                acc[i][j] = __builtin_amdgcn_wmma_f32_16x16x32_bf16(
                    false, af[i], false, bfr[j], (short)0, acc[i][j], false, false);

        if (more) {
            v4u* da = reinterpret_cast<v4u*>(&As[buf ^ 1][arow * LDS_STRIDE + akh]);
            da[0] = ra0; da[1] = ra1;
            *reinterpret_cast<v4u*>(&Bs[buf ^ 1][brow * LDS_STRIDE + bq]) = rb;
            buf ^= 1;
        }
        __syncthreads();
    }

    const int half = lane >> 4, ln = lane & 15;
    #pragma unroll
    for (int i = 0; i < 2; ++i) {
        const int mb = m0 + wm * 32 + i * 16 + half * 8;
        #pragma unroll
        for (int j = 0; j < 2; ++j) {
            const int nb = n0 + wn * 32 + j * 16 + ln;
            const float bv = bias[nb];
            #pragma unroll
            for (int v = 0; v < 8; ++v)
                h1[(long long)(mb + v) * C1 + nb] = acc[i][j][v] + bv;
        }
    }
}

// ---------------------------------------------------------------------------
// BatchNorm stats (training mode, biased var over rows): one block per channel.
// Produces fused scale/shift: y = x*scale + shift.
// ---------------------------------------------------------------------------
template <typename InT>
__global__ __launch_bounds__(256)
void bn_stats_kernel(const InT* __restrict__ h, int M, int N,
                     const float* __restrict__ gamma, const float* __restrict__ beta,
                     float* __restrict__ scale, float* __restrict__ shift)
{
    const int ch = blockIdx.x;
    float s = 0.f, s2 = 0.f;
    for (int r = threadIdx.x; r < M; r += 256) {
        const float v = (float)h[(long long)r * N + ch];
        s += v; s2 += v * v;
    }
    __shared__ float ss[256], ss2[256];
    ss[threadIdx.x] = s; ss2[threadIdx.x] = s2;
    __syncthreads();
    for (int o = 128; o > 0; o >>= 1) {
        if (threadIdx.x < o) {
            ss[threadIdx.x]  += ss[threadIdx.x + o];
            ss2[threadIdx.x] += ss2[threadIdx.x + o];
        }
        __syncthreads();
    }
    if (threadIdx.x == 0) {
        const float invM = 1.0f / (float)M;
        const float mean = ss[0] * invM;
        const float var  = ss2[0] * invM - mean * mean;
        const float sc   = gamma[ch] * rsqrtf(var + 1e-5f);
        scale[ch] = sc;
        shift[ch] = beta[ch] - mean * sc;
    }
}

// Apply BN1 affine and downconvert for the next WMMA GEMM.
__global__ void bn_apply_bf16_kernel(const float* __restrict__ h,
                                     const float* __restrict__ scale,
                                     const float* __restrict__ shift,
                                     __bf16* __restrict__ out, int chMask)
{
    long long i = (long long)blockIdx.x * blockDim.x + threadIdx.x;
    const int ch = (int)(i & chMask);
    out[i] = (__bf16)(h[i] * scale[ch] + shift[ch]);
}

// BN2 affine + softmax over the 1024-channel axis; one block (1024 thr) / pixel.
__global__ __launch_bounds__(1024)
void bn_softmax_kernel(const __bf16* __restrict__ kpre,
                       const float* __restrict__ scale,
                       const float* __restrict__ shift,
                       __bf16* __restrict__ kmap)
{
    const long long row = blockIdx.x;
    const int ch = threadIdx.x;
    const float v = (float)kpre[row * C2 + ch] * scale[ch] + shift[ch];
    __shared__ float red[1024];
    red[ch] = v;
    __syncthreads();
    for (int o = 512; o > 0; o >>= 1) {
        if (ch < o) red[ch] = fmaxf(red[ch], red[ch + o]);
        __syncthreads();
    }
    const float mx = red[0];
    __syncthreads();
    const float e = __expf(v - mx);
    red[ch] = e;
    __syncthreads();
    for (int o = 512; o > 0; o >>= 1) {
        if (ch < o) red[ch] += red[ch + o];
        __syncthreads();
    }
    kmap[row * C2 + ch] = (__bf16)(e / red[0]);
}

// ---------------------------------------------------------------------------
// Host-side orchestration
// ---------------------------------------------------------------------------
extern "C" void kernel_launch(void* const* d_in, const int* in_sizes, int n_in,
                              void* d_out, int out_size, void* d_ws, size_t ws_size,
                              hipStream_t stream) {
    const float* x       = (const float*)d_in[0];
    const float* conv1_w = (const float*)d_in[1];
    const float* conv1_b = (const float*)d_in[2];
    const float* bn1_g   = (const float*)d_in[3];
    const float* bn1_b   = (const float*)d_in[4];
    const float* conv2_w = (const float*)d_in[5];
    const float* conv2_b = (const float*)d_in[6];
    const float* bn2_g   = (const float*)d_in[7];
    const float* bn2_b   = (const float*)d_in[8];
    float* out = (float*)d_out;

    char* ws = (char*)d_ws;
    size_t off = 0;
    auto carve = [&](size_t bytes) {
        void* p = ws + off;
        off += (bytes + 255) & ~(size_t)255;
        return p;
    };
    __bf16* xT     = (__bf16*)carve((size_t)BATCH * HW * CIN * 2);   //  32 MB [b][hw][c]
    __bf16* xN     = (__bf16*)carve((size_t)BATCH * CIN * HW * 2);   //  32 MB [b][c][hw]
    __bf16* w1t    = (__bf16*)carve((size_t)49 * C1 * CIN * 2);      // 6.1 MB [tap][oc][c]
    __bf16* w2b    = (__bf16*)carve((size_t)C2 * CIN * 2);           // 0.5 MB [oc2][c]
    float*  h1     = (float*) carve((size_t)NPIX * C1 * 4);          //  64 MB
    __bf16* h1n    = (__bf16*)carve((size_t)NPIX * C1 * 2);          //  32 MB
    __bf16* kpre   = (__bf16*)carve((size_t)NPIX * C2 * 2);          // 128 MB
    __bf16* kmap   = (__bf16*)carve((size_t)NPIX * C2 * 2);          // 128 MB
    float*  scale1 = (float*) carve(C1 * 4);
    float*  shift1 = (float*) carve(C1 * 4);
    float*  scale2 = (float*) carve(C2 * 4);
    float*  shift2 = (float*) carve(C2 * 4);

    // 1) layout prep (bf16 converts; only conv1's operands (xT, w1t) need a transpose)
    prep_xT_kernel <<<(BATCH * HW * CIN) / 256, 256, 0, stream>>>(x, xT);
    cvt_bf16_kernel<<<(BATCH * CIN * HW) / 256, 256, 0, stream>>>(x, xN);
    prep_w1t_kernel<<<(49 * C1 * CIN) / 256,    256, 0, stream>>>(conv1_w, w1t);
    cvt_bf16_kernel<<<(C2 * CIN) / 256,         256, 0, stream>>>(conv2_w, w2b);

    // 2) conv1 (implicit GEMM, bf16 WMMA) -> h1 fp32 [pixel][oc]
    conv1_wmma_kernel<<<dim3(C1 / BN, NPIX / BM, 1), 256, 0, stream>>>(xT, w1t, conv1_b, h1);

    // 3) BN1 stats + apply (fold affine, downconvert for conv2's A)
    bn_stats_kernel<float><<<C1, 256, 0, stream>>>(h1, NPIX, C1, bn1_g, bn1_b, scale1, shift1);
    bn_apply_bf16_kernel<<<(NPIX * C1) / 256, 256, 0, stream>>>(h1, scale1, shift1, h1n, C1 - 1);

    // 4) conv2 1x1: [NPIX x 256] x [256 x 1024] -> kpre bf16 (Bt = w2b native layout)
    gemm_bf16_kernel<__bf16><<<dim3(C2 / BN, NPIX / BM, 1), 256, 0, stream>>>(
        h1n, w2b, kpre, conv2_b, NPIX, C2, CIN, 0LL, 0LL, 0LL);

    // 5) BN2 stats, then fused BN2-affine + softmax -> kmap bf16 [pixel][uv]
    bn_stats_kernel<__bf16><<<C2, 256, 0, stream>>>(kpre, NPIX, C2, bn2_g, bn2_b, scale2, shift2);
    bn_softmax_kernel<<<NPIX, 1024, 0, stream>>>(kpre, scale2, shift2, kmap);

    // 6) attention: per-batch GEMM  Y[ij][c] = kmap[ij][uv] x x[b][c][uv]^T
    //    (Bt = xN native layout). Output flat [b][ij][c] == reference reshape.
    gemm_bf16_kernel<float><<<dim3(CIN / BN, HW / BM, BATCH), 256, 0, stream>>>(
        kmap, xN, out, nullptr, HW, CIN, C2,
        (long long)HW * C2, (long long)HW * CIN, (long long)HW * CIN);
}
